// DMPNNEncoder_7361573945974
// MI455X (gfx1250) — compile-verified, matrix operand order checked by reading, not run on hardware
//
#include <hip/hip_runtime.h>
#include <hip/hip_bf16.h>

typedef __attribute__((ext_vector_type(16))) __bf16 v16bf;
typedef __attribute__((ext_vector_type(8)))  float  v8f;

union Frag { uint4 q[2]; v16bf v; };

__device__ __forceinline__ unsigned int pack_bf16(float a, float b) {
    unsigned int ua = __float_as_uint(a);
    unsigned int ub = __float_as_uint(b);
    ua += 0x7FFFu + ((ua >> 16) & 1u);   // round-to-nearest-even
    ub += 0x7FFFu + ((ub >> 16) & 1u);
    return (ua >> 16) | (ub & 0xFFFF0000u);
}

__device__ __forceinline__ v8f wmma_bf16(const v16bf& a, const v16bf& b, v8f c) {
    return __builtin_amdgcn_wmma_f32_16x16x32_bf16(false, a, false, b, (short)0, c, false, false);
}

// ---------------------------------------------------------------------------
// H0 = relu([X[src]; B] @ W_i^T) ; H = H0.   K = 144 padded to 160.
// One wave per 16-edge tile; 4 waves / block; W_i (bf16, K-padded) in LDS.
// ---------------------------------------------------------------------------
__global__ __launch_bounds__(128) void h0_kernel(
    const float* __restrict__ X, const float* __restrict__ B,
    const int* __restrict__ src, const float* __restrict__ W_i,
    float* __restrict__ H0, float* __restrict__ H, int E)
{
    __shared__ __align__(16) unsigned int wLds[128 * 80];     // 40 KB
    __shared__ __align__(16) unsigned int aLds[4 * 16 * 80];  // 20 KB

    const int tid  = threadIdx.x;
    const int lane = tid & 31;
    const int wid  = tid >> 5;

    // cooperative weight load: W_i[n][k], k padded 144->160 with zeros
    for (int idx = tid; idx < 128 * 80; idx += 128) {
        int n  = idx / 80;
        int k0 = (idx % 80) * 2;
        float a = (k0     < 144) ? W_i[n * 144 + k0]     : 0.0f;
        float b = (k0 + 1 < 144) ? W_i[n * 144 + k0 + 1] : 0.0f;
        wLds[idx] = pack_bf16(a, b);
    }
    __syncthreads();

    const int tile = blockIdx.x * 4 + wid;
    const int tilebase = tile * 16;
    if (tilebase >= E) return;
    const bool full = (tilebase + 16 <= E);

    unsigned int* aw = aLds + wid * 16 * 80;

    // stage A tile = [X[src[e]] (128) ; B[e] (16) ; zeros (16)] as bf16
    for (int m = 0; m < 16; ++m) {
        int e = tilebase + m; if (e >= E) e = E - 1;
        int s = src[e];
        const float4 xv = *(const float4*)(X + (size_t)s * 128 + lane * 4);
        aw[m * 80 + lane * 2]     = pack_bf16(xv.x, xv.y);
        aw[m * 80 + lane * 2 + 1] = pack_bf16(xv.z, xv.w);
        if (lane < 4) {
            const float4 bv = *(const float4*)(B + (size_t)e * 16 + lane * 4);
            aw[m * 80 + 64 + lane * 2]     = pack_bf16(bv.x, bv.y);
            aw[m * 80 + 64 + lane * 2 + 1] = pack_bf16(bv.z, bv.w);
        } else if (lane < 8) {
            aw[m * 80 + 64 + lane * 2]     = 0u;
            aw[m * 80 + 64 + lane * 2 + 1] = 0u;
        }
    }

    const int m_ = lane & 15, kh = lane >> 4;
    Frag af[5];
#pragma unroll
    for (int kc = 0; kc < 5; ++kc) {
        int base = m_ * 80 + kc * 16 + kh * 4;
        af[kc].q[0] = *(const uint4*)(aw + base);
        af[kc].q[1] = *(const uint4*)(aw + base + 8);
    }

    const size_t rowbase = (size_t)(tilebase + 8 * kh) * 128;

#pragma unroll
    for (int nt = 0; nt < 8; ++nt) {
        v8f acc = {0.f, 0.f, 0.f, 0.f, 0.f, 0.f, 0.f, 0.f};
#pragma unroll
        for (int kc = 0; kc < 5; ++kc) {
            Frag bf_;
            int bbase = (nt * 16 + m_) * 80 + kc * 16 + kh * 4;
            bf_.q[0] = *(const uint4*)(wLds + bbase);
            bf_.q[1] = *(const uint4*)(wLds + bbase + 8);
            acc = wmma_bf16(af[kc].v, bf_.v, acc);
        }
        const int gcol = nt * 16 + m_;
        float* __restrict__ h0p = H0 + rowbase + gcol;
        float* __restrict__ hp  = H  + rowbase + gcol;
        if (full) {
#pragma unroll
            for (int r = 0; r < 8; ++r) {
                float v = fmaxf(acc[r], 0.f);
                h0p[(size_t)r * 128] = v;
                hp[(size_t)r * 128]  = v;
            }
        } else {
#pragma unroll
            for (int r = 0; r < 8; ++r) {
                if (tilebase + r + 8 * kh < E) {
                    float v = fmaxf(acc[r], 0.f);
                    h0p[(size_t)r * 128] = v;
                    hp[(size_t)r * 128]  = v;
                }
            }
        }
    }
}

// ---------------------------------------------------------------------------
// H = relu(H0 + (node_sum[src] - H[rev]) @ W_h^T), in place.
// rev = e^1 stays inside each 16-aligned tile -> no cross-block hazard.
// ---------------------------------------------------------------------------
__global__ __launch_bounds__(128) void step_kernel(
    const float* __restrict__ node_sum, const int* __restrict__ src,
    const int* __restrict__ rev, const float* __restrict__ W_h,
    const float* __restrict__ H0, float* __restrict__ H, int E)
{
    __shared__ __align__(16) unsigned int wLds[128 * 64];     // 32 KB
    __shared__ __align__(16) unsigned int aLds[4 * 16 * 64];  // 16 KB

    const int tid  = threadIdx.x;
    const int lane = tid & 31;
    const int wid  = tid >> 5;

    for (int idx = tid; idx < 128 * 64; idx += 128) {
        int n = idx >> 6, k0 = (idx & 63) * 2;
        wLds[idx] = pack_bf16(W_h[n * 128 + k0], W_h[n * 128 + k0 + 1]);
    }
    __syncthreads();

    const int tile = blockIdx.x * 4 + wid;
    const int tilebase = tile * 16;
    if (tilebase >= E) return;
    const bool full = (tilebase + 16 <= E);

    unsigned int* aw = aLds + wid * 16 * 64;

    // stage A tile = node_sum[src[e]] - H[rev[e]]  (bf16)
    for (int m = 0; m < 16; ++m) {
        int e = tilebase + m; if (e >= E) e = E - 1;
        int s = src[e];
        int r = rev[e];
        const float4 nv = *(const float4*)(node_sum + (size_t)s * 128 + lane * 4);
        const float4 hv = *(const float4*)(H        + (size_t)r * 128 + lane * 4);
        aw[m * 64 + lane * 2]     = pack_bf16(nv.x - hv.x, nv.y - hv.y);
        aw[m * 64 + lane * 2 + 1] = pack_bf16(nv.z - hv.z, nv.w - hv.w);
    }

    const int m_ = lane & 15, kh = lane >> 4;
    Frag af[4];
#pragma unroll
    for (int kc = 0; kc < 4; ++kc) {
        int base = m_ * 64 + kc * 16 + kh * 4;
        af[kc].q[0] = *(const uint4*)(aw + base);
        af[kc].q[1] = *(const uint4*)(aw + base + 8);
    }

    const size_t rowbase = (size_t)(tilebase + 8 * kh) * 128;

#pragma unroll
    for (int nt = 0; nt < 8; ++nt) {
        v8f acc = {0.f, 0.f, 0.f, 0.f, 0.f, 0.f, 0.f, 0.f};
#pragma unroll
        for (int kc = 0; kc < 4; ++kc) {
            Frag bf_;
            int bbase = (nt * 16 + m_) * 64 + kc * 16 + kh * 4;
            bf_.q[0] = *(const uint4*)(wLds + bbase);
            bf_.q[1] = *(const uint4*)(wLds + bbase + 8);
            acc = wmma_bf16(af[kc].v, bf_.v, acc);
        }
        const int gcol = nt * 16 + m_;
        const float* __restrict__ h0p = H0 + rowbase + gcol;
        float* __restrict__ hp = H + rowbase + gcol;
        if (full) {
#pragma unroll
            for (int r = 0; r < 8; ++r) {
                float v = h0p[(size_t)r * 128] + acc[r];
                hp[(size_t)r * 128] = fmaxf(v, 0.f);
            }
        } else {
#pragma unroll
            for (int r = 0; r < 8; ++r) {
                if (tilebase + r + 8 * kh < E) {
                    float v = h0p[(size_t)r * 128] + acc[r];
                    hp[(size_t)r * 128] = fmaxf(v, 0.f);
                }
            }
        }
    }
}

// ---------------------------------------------------------------------------
// P = relu([X ; node_sum] @ W_a^T); Z_G += P per graph (fused atomic pool).
// K = 256 split into two 128-halves so static LDS stays at 48 KB.
// ---------------------------------------------------------------------------
__global__ __launch_bounds__(128) void readout_kernel(
    const float* __restrict__ X, const float* __restrict__ node_sum,
    const float* __restrict__ W_a, const int* __restrict__ batch,
    float* __restrict__ out, int n_nodes)
{
    __shared__ __align__(16) unsigned int wLds[128 * 64];     // 32 KB
    __shared__ __align__(16) unsigned int aLds[4 * 16 * 64];  // 16 KB

    const int tid  = threadIdx.x;
    const int lane = tid & 31;
    const int wid  = tid >> 5;
    const int tile = blockIdx.x * 4 + wid;
    const int tilebase = tile * 16;
    const bool active = tilebase < n_nodes;
    const int m_ = lane & 15, kh = lane >> 4;

    v8f acc[8];
#pragma unroll
    for (int nt = 0; nt < 8; ++nt)
        acc[nt] = (v8f){0.f, 0.f, 0.f, 0.f, 0.f, 0.f, 0.f, 0.f};

    unsigned int* aw = aLds + wid * 16 * 64;

    for (int half = 0; half < 2; ++half) {
        __syncthreads();  // protect previous half's LDS reads
        for (int idx = tid; idx < 128 * 64; idx += 128) {
            int n = idx >> 6, k0 = half * 128 + (idx & 63) * 2;
            wLds[idx] = pack_bf16(W_a[n * 256 + k0], W_a[n * 256 + k0 + 1]);
        }
        __syncthreads();

        if (active) {
            const float* srcmat = (half == 0) ? X : node_sum;
            for (int m = 0; m < 16; ++m) {
                int grow = tilebase + m; if (grow >= n_nodes) grow = n_nodes - 1;
                const float4 v = *(const float4*)(srcmat + (size_t)grow * 128 + lane * 4);
                aw[m * 64 + lane * 2]     = pack_bf16(v.x, v.y);
                aw[m * 64 + lane * 2 + 1] = pack_bf16(v.z, v.w);
            }
            Frag af[4];
#pragma unroll
            for (int kc = 0; kc < 4; ++kc) {
                int base = m_ * 64 + kc * 16 + kh * 4;
                af[kc].q[0] = *(const uint4*)(aw + base);
                af[kc].q[1] = *(const uint4*)(aw + base + 8);
            }
#pragma unroll
            for (int nt = 0; nt < 8; ++nt) {
#pragma unroll
                for (int kc = 0; kc < 4; ++kc) {
                    Frag bf_;
                    int bbase = (nt * 16 + m_) * 64 + kc * 16 + kh * 4;
                    bf_.q[0] = *(const uint4*)(wLds + bbase);
                    bf_.q[1] = *(const uint4*)(wLds + bbase + 8);
                    acc[nt] = wmma_bf16(af[kc].v, bf_.v, acc[nt]);
                }
            }
        }
    }

    if (active) {
        // load batch/graph index once per destination row (8 per lane)
        int brow[8];
#pragma unroll
        for (int r = 0; r < 8; ++r) {
            int grow = tilebase + r + 8 * kh;
            brow[r] = (grow < n_nodes) ? batch[grow] : -1;
        }
#pragma unroll
        for (int nt = 0; nt < 8; ++nt) {
            const int gcol = nt * 16 + m_;
#pragma unroll
            for (int r = 0; r < 8; ++r) {
                if (brow[r] >= 0) {
                    float v = fmaxf(acc[nt][r], 0.f);
                    atomicAdd(out + (size_t)brow[r] * 128 + gcol, v);
                }
            }
        }
    }
}

// ---------------------------------------------------------------------------
// node_sum[rcv[e]] += H[e]   (segment_sum via f32 atomics, float4 per thread)
// ---------------------------------------------------------------------------
__global__ void scatter_kernel(const float* __restrict__ H,
                               const int* __restrict__ rcv,
                               float* __restrict__ node_sum, int E)
{
    long long idx = (long long)blockIdx.x * blockDim.x + threadIdx.x;
    int e = (int)(idx >> 5);
    if (e >= E) return;
    int c = (int)(idx & 31) * 4;
    int r = rcv[e];
    const float4 h = *(const float4*)(H + (size_t)e * 128 + c);
    float* base = node_sum + (size_t)r * 128 + c;
    atomicAdd(base + 0, h.x);
    atomicAdd(base + 1, h.y);
    atomicAdd(base + 2, h.z);
    atomicAdd(base + 3, h.w);
}

__global__ void zero_kernel(float* __restrict__ p, long long n)
{
    long long i = ((long long)blockIdx.x * blockDim.x + threadIdx.x) * 4;
    if (i + 3 < n) {
        *(float4*)(p + i) = make_float4(0.f, 0.f, 0.f, 0.f);
    } else {
        for (long long j = i; j < n; ++j) p[j] = 0.f;
    }
}

// ---------------------------------------------------------------------------
extern "C" void kernel_launch(void* const* d_in, const int* in_sizes, int n_in,
                              void* d_out, int out_size, void* d_ws, size_t ws_size,
                              hipStream_t stream)
{
    const float* X          = (const float*)d_in[0];
    const float* B          = (const float*)d_in[1];
    const int*   edge_index = (const int*)  d_in[2];
    const int*   rev        = (const int*)  d_in[3];
    const int*   batch      = (const int*)  d_in[4];
    const float* W_i        = (const float*)d_in[6];
    const float* W_h        = (const float*)d_in[7];
    const float* W_a        = (const float*)d_in[8];
    float* out = (float*)d_out;

    const int n_nodes = in_sizes[0] / 128;
    const int E       = in_sizes[1] / 16;

    char* w = (char*)d_ws;
    const size_t szH = (size_t)E * 128 * sizeof(float);
    float* H0       = (float*)w;
    float* H        = (float*)(w + szH);
    float* node_sum = (float*)(w + 2 * szH);

    const int* src = edge_index;
    const int* rcv = edge_index + E;

    const int etiles  = (E + 15) / 16;
    const int eblocks = (etiles + 3) / 4;
    const long long nsElems = (long long)n_nodes * 128;
    const int zblocksNS  = (int)((nsElems / 4 + 255) / 256);
    const int scblocks   = (int)(((long long)E * 32 + 255) / 256);

    h0_kernel<<<eblocks, 128, 0, stream>>>(X, B, src, W_i, H0, H, E);

    for (int step = 0; step < 5; ++step) {
        zero_kernel<<<zblocksNS, 256, 0, stream>>>(node_sum, nsElems);
        scatter_kernel<<<scblocks, 256, 0, stream>>>(H, rcv, node_sum, E);
        step_kernel<<<eblocks, 128, 0, stream>>>(node_sum, src, rev, W_h, H0, H, E);
    }

    zero_kernel<<<zblocksNS, 256, 0, stream>>>(node_sum, nsElems);
    scatter_kernel<<<scblocks, 256, 0, stream>>>(H, rcv, node_sum, E);

    const long long outElems = (long long)out_size;
    zero_kernel<<<(int)((outElems / 4 + 255) / 256), 256, 0, stream>>>(out, outElems);

    const int ntiles  = (n_nodes + 15) / 16;
    const int nblocks = (ntiles + 3) / 4;
    readout_kernel<<<nblocks, 128, 0, stream>>>(X, node_sum, W_a, batch, out, n_nodes);
}
